// GATNet_38714835206274
// MI455X (gfx1250) — matrix operand order repaired and checked by reading, not compiled, not run on hardware
//
#include <hip/hip_runtime.h>
#include <hip/hip_bf16.h>

// ---------------------------------------------------------------------------
// GATNet forward for MI455X (gfx1250): two GAT layers + mean-pool + log_softmax
// Dense projections via v_wmma_f32_16x16x32_bf16; edge softmax via L2 atomics.
// ---------------------------------------------------------------------------

#define NN_NODES 50000
#define NE_EDGES 800000
#define D_IN     256
#define D_HID    32
#define N_HEADS  8
#define N_CLASS  10
#define N_GRAPHS 64
#define ALPHA_LR 0.2f

typedef __attribute__((ext_vector_type(16))) __bf16 v16bf;
typedef __attribute__((ext_vector_type(8)))  float  v8f;

union FragU {
    v16bf v;
    unsigned int u[8];
};

__device__ __forceinline__ unsigned short f32_to_bf16_rne(float f) {
    unsigned int u = __float_as_uint(f);
    unsigned int r = (u + 0x7FFFu + ((u >> 16) & 1u)) >> 16;
    return (unsigned short)r;
}

__device__ __forceinline__ unsigned int pack_bf16x2(float lo, float hi) {
    return (unsigned int)f32_to_bf16_rne(lo) |
           ((unsigned int)f32_to_bf16_rne(hi) << 16);
}

__device__ __forceinline__ void atomicMaxF(float* addr, float val) {
    if (val >= 0.0f)
        atomicMax((int*)addr, __float_as_int(val));
    else
        atomicMin((unsigned int*)addr, __float_as_uint(val));
}

// ---------------------------------------------------------------------------
// WMMA GEMM: C[M,N] = A[M,K] * B[K,N], fp32 in/out, bf16 WMMA inside.
// Block = 128 threads (4 waves). Block tile = 64 rows x NT*16 cols (NT is a
// compile-time template param so the accumulator dataflow has NO conditionals).
// Wave w owns rows [w*16, w*16+16); all NT column sub-tiles per wave.
// K-step = 64 (two WMMA K-slices per LDS staging round).
// LDS holds packed bf16x2 dwords arranged so each lane's WMMA fragment is
// contiguous: every fragment = 2x ds_load_b128. K % 64 == 0 required.
// ---------------------------------------------------------------------------
#define GEMM_BLOCK 128
#define LDS_STRIDE 36   // dwords/row: 144B = 9*16B (aligned b128), 36m%64 distinct

template <int NT>
__global__ __launch_bounds__(GEMM_BLOCK)
void wmma_gemm_bf16_kernel(const float* __restrict__ A,
                           const float* __restrict__ B,
                           float* __restrict__ C,
                           int M, int K, int N) {
    // Apair[r][kp]: dword = bf16(A[r][2kp]) | bf16(A[r][2kp+1])<<16, r in [0,64)
    // Bpair[n][kp]: dword = bf16(B[2kp][n]) | bf16(B[2kp+1][n])<<16, n in [0,NT*16)
    __shared__ unsigned int Apair[64 * LDS_STRIDE];
    __shared__ unsigned int Bpair[NT * 16 * LDS_STRIDE];

    const int tid  = threadIdx.x;
    const int lane = tid & 31;
    const int wave = tid >> 5;
    const int row0 = blockIdx.x * 64;
    const int colBase = blockIdx.y * (NT * 16);

    const int l16 = lane & 15;   // row (A) / col (B,C) within 16-tile
    const int hf  = lane >> 4;   // lane half selects K sub-group / C row offset

    v8f acc[NT];
#pragma unroll
    for (int t = 0; t < NT; ++t) acc[t] = (v8f){};

    for (int k0 = 0; k0 < K; k0 += 64) {
        // ---- stage A: 64 rows x 64 k. 1024 float4 loads -> packed dwords ----
        // Row index is CLAMPED (not branched): padded rows hold garbage that is
        // never stored (store is row-guarded). Keeps staging branch-free.
#pragma unroll
        for (int i = 0; i < 8; ++i) {
            int idx = tid + i * GEMM_BLOCK;       // 0..1023
            int r = idx >> 4, q = idx & 15;       // q: float4 index along k
            int gr = row0 + r;
            gr = (gr < M) ? gr : (M - 1);
            float4 f4 = *(const float4*)(A + (size_t)gr * K + k0 + q * 4);
            uint2 pk = make_uint2(pack_bf16x2(f4.x, f4.y), pack_bf16x2(f4.z, f4.w));
            *(uint2*)&Apair[r * LDS_STRIDE + q * 2] = pk;
        }
        // ---- stage B: 64 k x NT*16 n -> transposed packed dwords ----
#pragma unroll
        for (int i = 0; i < NT * 4; ++i) {
            int idx = tid + i * GEMM_BLOCK;       // 0..NT*512-1
            int qp = idx / (NT * 16);             // kpair 0..31
            int n  = idx % (NT * 16);             // consecutive lanes -> consecutive n
            int gn = colBase + n;
            float b0 = 0.f, b1 = 0.f;
            if (gn < N) {
                b0 = B[(size_t)(k0 + 2 * qp) * N + gn];
                b1 = B[(size_t)(k0 + 2 * qp + 1) * N + gn];
            }
            Bpair[n * LDS_STRIDE + qp] = pack_bf16x2(b0, b1);
        }
        __syncthreads();

        // ---- two K-slices of 32, NT tiles each: all WMMAs unconditional ----
#pragma unroll
        for (int s = 0; s < 2; ++s) {
            // A fragment (ISA 7.12.2 16-bit A 16x32):
            // VGPR v<4: kpair = s*16 + hf*4 + v ; v>=4: kpair = s*16 + 8 + hf*4 + (v-4)
            FragU fa;
            {
                const unsigned int* ap =
                    &Apair[(wave * 16 + l16) * LDS_STRIDE + s * 16 + hf * 4];
                uint4 a0 = *(const uint4*)ap;
                uint4 a1 = *(const uint4*)(ap + 8);
                fa.u[0] = a0.x; fa.u[1] = a0.y; fa.u[2] = a0.z; fa.u[3] = a0.w;
                fa.u[4] = a1.x; fa.u[5] = a1.y; fa.u[6] = a1.z; fa.u[7] = a1.w;
            }
#pragma unroll
            for (int t = 0; t < NT; ++t) {
                // B fragment: VGPR v<4: kpair = s*16 + hf*4 + v ; v>=4: kpair = s*16 + 8 + hf*4 + (v-4)
                const unsigned int* bp =
                    &Bpair[(t * 16 + l16) * LDS_STRIDE + s * 16 + hf * 4];
                uint4 b0 = *(const uint4*)bp;
                uint4 b1 = *(const uint4*)(bp + 8);
                FragU fb;
                fb.u[0] = b0.x; fb.u[1] = b0.y; fb.u[2] = b0.z; fb.u[3] = b0.w;
                fb.u[4] = b1.x; fb.u[5] = b1.y; fb.u[6] = b1.z; fb.u[7] = b1.w;
                acc[t] = __builtin_amdgcn_wmma_f32_16x16x32_bf16(
                    false, fa.v, false, fb.v, (short)0, acc[t], false, false);
            }
        }
        __syncthreads();
    }

    // ---- store D: VGPR v -> row (wave*16 + hf*8 + v), col = lane%16 ----
#pragma unroll
    for (int t = 0; t < NT; ++t) {
        int gc = colBase + t * 16 + l16;
        if (gc < N) {
#pragma unroll
            for (int v = 0; v < 8; ++v) {
                int gr = row0 + wave * 16 + hf * 8 + v;
                if (gr < M) C[(size_t)gr * N + gc] = acc[t][v];
            }
        }
    }
}

// ---------------------------------------------------------------------------
// Elementwise / attention kernels
// ---------------------------------------------------------------------------
__global__ void fill_f32_kernel(float* __restrict__ p, int n, float v) {
    int t = blockIdx.x * blockDim.x + threadIdx.x;
    if (t < n) p[t] = v;
}

// e_src[n,h] = sum_j h[n,h,j]*a[h,j]; e_dst[n,h] = sum_j h[n,h,j]*a[h,d+j]
__global__ void attn_coef_kernel(const float* __restrict__ h,
                                 const float* __restrict__ a,
                                 float* __restrict__ esrc,
                                 float* __restrict__ edst,
                                 int Nn, int H, int d) {
    int t = blockIdx.x * blockDim.x + threadIdx.x;
    if (t >= Nn * H) return;
    int hh = t % H, n = t / H;
    const float* hp = h + (size_t)n * H * d + (size_t)hh * d;
    const float* ap = a + (size_t)hh * 2 * d;
    float s1 = 0.f, s2 = 0.f;
    for (int j = 0; j < d; ++j) {
        float v = hp[j];
        s1 += v * ap[j];
        s2 += v * ap[d + j];
    }
    esrc[t] = s1;
    edst[t] = s2;
}

// e = leaky_relu(esrc[src]+edst[dst]); segment max into m[dst]
__global__ void edge_e_max_kernel(const int* __restrict__ src,
                                  const int* __restrict__ dst,
                                  const float* __restrict__ esrc,
                                  const float* __restrict__ edst,
                                  float* __restrict__ e,
                                  float* __restrict__ m,
                                  int E, int H) {
    int t = blockIdx.x * blockDim.x + threadIdx.x;
    if (t >= E * H) return;
    int hh = t % H, ed = t / H;
    int s = src[ed], dd = dst[ed];
    float v = esrc[s * H + hh] + edst[dd * H + hh];
    v = (v > 0.f) ? v : ALPHA_LR * v;
    e[t] = v;
    atomicMaxF(&m[dd * H + hh], v);
}

// w = exp(e - m[dst]); segment sum into z[dst]; w overwrites e
__global__ void edge_w_sum_kernel(const int* __restrict__ dst,
                                  float* __restrict__ e,
                                  const float* __restrict__ m,
                                  float* __restrict__ z,
                                  int E, int H) {
    int t = blockIdx.x * blockDim.x + threadIdx.x;
    if (t >= E * H) return;
    int hh = t % H, ed = t / H;
    int dd = dst[ed];
    float w = expf(e[t] - m[dd * H + hh]);
    e[t] = w;
    atomicAdd(&z[dd * H + hh], w);
}

// agg[dst,h,j] += (w/z[dst,h]) * hfeat[src,h,j]  -- lane-consecutive j => coalesced
__global__ void edge_msg_kernel(const int* __restrict__ src,
                                const int* __restrict__ dst,
                                const float* __restrict__ w,
                                const float* __restrict__ z,
                                const float* __restrict__ hfeat,
                                float* __restrict__ agg,
                                int E, int H, int d) {
    int t = blockIdx.x * blockDim.x + threadIdx.x;
    if (t >= E * H * d) return;
    int j = t % d;
    int rem = t / d;
    int hh = rem % H;
    int ed = rem / H;
    int s = src[ed], dd = dst[ed];
    float att = w[ed * H + hh] / (z[dd * H + hh] + 1e-16f);
    float val = att * hfeat[((size_t)s * H + hh) * d + j];
    atomicAdd(&agg[((size_t)dd * H + hh) * d + j], val);
}

__global__ void elu_inplace_kernel(float* __restrict__ p, int n) {
    int t = blockIdx.x * blockDim.x + threadIdx.x;
    if (t >= n) return;
    float v = p[t];
    p[t] = (v > 0.f) ? v : (expf(v) - 1.0f);
}

// scatter node features into per-graph sums + counts
__global__ void pool_scatter_kernel(const int* __restrict__ batch,
                                    const float* __restrict__ h2,
                                    float* __restrict__ sums,
                                    float* __restrict__ cnts,
                                    int Nn, int C) {
    int t = blockIdx.x * blockDim.x + threadIdx.x;
    if (t >= Nn * C) return;
    int c = t % C, n = t / C;
    int g = batch[n];
    atomicAdd(&sums[g * C + c], h2[t]);
    if (c == 0) atomicAdd(&cnts[g], 1.0f);
}

__global__ void pool_finish_kernel(const float* __restrict__ sums,
                                   const float* __restrict__ cnts,
                                   float* __restrict__ out,
                                   int G, int C) {
    int g = blockIdx.x * blockDim.x + threadIdx.x;
    if (g >= G) return;
    float c = cnts[g];
    c = (c > 1.0f) ? c : 1.0f;
    float v[N_CLASS];
    float mx = -3.0e38f;
    for (int i = 0; i < C; ++i) {
        v[i] = sums[g * C + i] / c;
        mx = (v[i] > mx) ? v[i] : mx;
    }
    float se = 0.f;
    for (int i = 0; i < C; ++i) se += expf(v[i] - mx);
    float lse = mx + logf(se);
    for (int i = 0; i < C; ++i) out[g * C + i] = v[i] - lse;
}

// ---------------------------------------------------------------------------
// Host-side orchestration
// ---------------------------------------------------------------------------
static inline int cdiv(long long a, long long b) { return (int)((a + b - 1) / b); }

extern "C" void kernel_launch(void* const* d_in, const int* in_sizes, int n_in,
                              void* d_out, int out_size, void* d_ws, size_t ws_size,
                              hipStream_t stream) {
    const float* x   = (const float*)d_in[0];  // [NN, D_IN]
    const int*   ei  = (const int*)d_in[1];    // [2, NE]
    const int*   bat = (const int*)d_in[2];    // [NN]
    const float* W1  = (const float*)d_in[3];  // [D_IN, H*DH]
    const float* a1  = (const float*)d_in[4];  // [H, 2*DH]
    const float* W2  = (const float*)d_in[5];  // [H*DH, N_CLASS]
    const float* a2  = (const float*)d_in[6];  // [1, 2*N_CLASS]
    float* out = (float*)d_out;                // [N_GRAPHS, N_CLASS]

    const int* src = ei;              // row 0
    const int* dst = ei + NE_EDGES;   // row 1

    // ---- workspace carve-up (floats) ----
    float* p = (float*)d_ws;
    float* h1raw = p; p += (size_t)NN_NODES * D_IN;        // pre-agg features L1
    float* h1agg = p; p += (size_t)NN_NODES * D_IN;        // aggregated L1 (ELU in place)
    float* esrc1 = p; p += (size_t)NN_NODES * N_HEADS;
    float* edst1 = p; p += (size_t)NN_NODES * N_HEADS;
    float* e1    = p; p += (size_t)NE_EDGES * N_HEADS;     // e then w
    float* m1    = p; p += (size_t)NN_NODES * N_HEADS;
    float* z1    = p; p += (size_t)NN_NODES * N_HEADS;
    float* h2raw = p; p += (size_t)NN_NODES * N_CLASS;
    float* h2agg = p; p += (size_t)NN_NODES * N_CLASS;
    float* esrc2 = p; p += (size_t)NN_NODES;
    float* edst2 = p; p += (size_t)NN_NODES;
    float* e2    = p; p += (size_t)NE_EDGES;
    float* m2    = p; p += (size_t)NN_NODES;
    float* z2    = p; p += (size_t)NN_NODES;
    float* sums  = p; p += (size_t)N_GRAPHS * N_CLASS;
    float* cnts  = p; p += (size_t)N_GRAPHS;

    const int TB = 256;

    // ===================== Layer 1 =====================
    // h1raw = x @ W1   (50000x256 * 256x256), WMMA bf16, NT=4 (N=256 exact)
    {
        dim3 grid(cdiv(NN_NODES, 64), (N_HEADS * D_HID) / 64);
        hipLaunchKernelGGL((wmma_gemm_bf16_kernel<4>), grid, dim3(GEMM_BLOCK), 0, stream,
                           x, W1, h1raw, NN_NODES, D_IN, N_HEADS * D_HID);
    }
    hipLaunchKernelGGL(attn_coef_kernel, dim3(cdiv((size_t)NN_NODES * N_HEADS, TB)), dim3(TB), 0, stream,
                       h1raw, a1, esrc1, edst1, NN_NODES, N_HEADS, D_HID);
    hipLaunchKernelGGL(fill_f32_kernel, dim3(cdiv((size_t)NN_NODES * N_HEADS, TB)), dim3(TB), 0, stream,
                       m1, NN_NODES * N_HEADS, -3.0e38f);
    hipLaunchKernelGGL(fill_f32_kernel, dim3(cdiv((size_t)NN_NODES * N_HEADS, TB)), dim3(TB), 0, stream,
                       z1, NN_NODES * N_HEADS, 0.0f);
    hipLaunchKernelGGL(fill_f32_kernel, dim3(cdiv((size_t)NN_NODES * D_IN, TB)), dim3(TB), 0, stream,
                       h1agg, NN_NODES * D_IN, 0.0f);
    hipLaunchKernelGGL(edge_e_max_kernel, dim3(cdiv((size_t)NE_EDGES * N_HEADS, TB)), dim3(TB), 0, stream,
                       src, dst, esrc1, edst1, e1, m1, NE_EDGES, N_HEADS);
    hipLaunchKernelGGL(edge_w_sum_kernel, dim3(cdiv((size_t)NE_EDGES * N_HEADS, TB)), dim3(TB), 0, stream,
                       dst, e1, m1, z1, NE_EDGES, N_HEADS);
    hipLaunchKernelGGL(edge_msg_kernel, dim3(cdiv((size_t)NE_EDGES * N_HEADS * D_HID, TB)), dim3(TB), 0, stream,
                       src, dst, e1, z1, h1raw, h1agg, NE_EDGES, N_HEADS, D_HID);
    hipLaunchKernelGGL(elu_inplace_kernel, dim3(cdiv((size_t)NN_NODES * D_IN, TB)), dim3(TB), 0, stream,
                       h1agg, NN_NODES * D_IN);

    // ===================== Layer 2 =====================
    // h2raw = h1act @ W2  (50000x256 * 256x10), WMMA bf16, NT=1 (cols 0..15, store<10)
    {
        dim3 grid(cdiv(NN_NODES, 64), 1);
        hipLaunchKernelGGL((wmma_gemm_bf16_kernel<1>), grid, dim3(GEMM_BLOCK), 0, stream,
                           h1agg, W2, h2raw, NN_NODES, N_HEADS * D_HID, N_CLASS);
    }
    hipLaunchKernelGGL(attn_coef_kernel, dim3(cdiv(NN_NODES, TB)), dim3(TB), 0, stream,
                       h2raw, a2, esrc2, edst2, NN_NODES, 1, N_CLASS);
    hipLaunchKernelGGL(fill_f32_kernel, dim3(cdiv(NN_NODES, TB)), dim3(TB), 0, stream,
                       m2, NN_NODES, -3.0e38f);
    hipLaunchKernelGGL(fill_f32_kernel, dim3(cdiv(NN_NODES, TB)), dim3(TB), 0, stream,
                       z2, NN_NODES, 0.0f);
    hipLaunchKernelGGL(fill_f32_kernel, dim3(cdiv((size_t)NN_NODES * N_CLASS, TB)), dim3(TB), 0, stream,
                       h2agg, NN_NODES * N_CLASS, 0.0f);
    hipLaunchKernelGGL(edge_e_max_kernel, dim3(cdiv(NE_EDGES, TB)), dim3(TB), 0, stream,
                       src, dst, esrc2, edst2, e2, m2, NE_EDGES, 1);
    hipLaunchKernelGGL(edge_w_sum_kernel, dim3(cdiv(NE_EDGES, TB)), dim3(TB), 0, stream,
                       dst, e2, m2, z2, NE_EDGES, 1);
    hipLaunchKernelGGL(edge_msg_kernel, dim3(cdiv((size_t)NE_EDGES * N_CLASS, TB)), dim3(TB), 0, stream,
                       src, dst, e2, z2, h2raw, h2agg, NE_EDGES, 1, N_CLASS);

    // ===================== Pool + log_softmax =====================
    hipLaunchKernelGGL(fill_f32_kernel, dim3(1), dim3(TB), 0, stream,
                       sums, N_GRAPHS * N_CLASS, 0.0f);
    hipLaunchKernelGGL(fill_f32_kernel, dim3(1), dim3(TB), 0, stream,
                       cnts, N_GRAPHS, 0.0f);
    hipLaunchKernelGGL(pool_scatter_kernel, dim3(cdiv((size_t)NN_NODES * N_CLASS, TB)), dim3(TB), 0, stream,
                       bat, h2agg, sums, cnts, NN_NODES, N_CLASS);
    hipLaunchKernelGGL(pool_finish_kernel, dim3(1), dim3(N_GRAPHS), 0, stream,
                       sums, cnts, out, N_GRAPHS, N_CLASS);
    (void)in_sizes; (void)n_in; (void)out_size; (void)ws_size;
}
